// DKL_45810121179236
// MI455X (gfx1250) — compile-verified, matrix operand order checked by reading, not run on hardware
//
#include <hip/hip_runtime.h>

typedef _Float16 v16h __attribute__((ext_vector_type(16)));
typedef _Float16 h8   __attribute__((ext_vector_type(8)));
typedef float    v8f  __attribute__((ext_vector_type(8)));

#define NROWS 8192
#define KDIM  256
#define BLK   128          // block tile: 128(M) x 128(N)
#define KCH   64           // K chunk staged per LDS buffer
#define NCHUNK (KDIM / KCH)
#define ROWH  72           // padded LDS row stride in halves (144 B, conflict-free)
#define BUFB  (2 * BLK * ROWH * 2)  // bytes per LDS buffer: 256 rows * 144 B = 36864
#define LDSB  (2 * BUFB)            // double buffered: 73728 B

// ---- order-preserving float <-> uint encoding (native u32 atomicMax) ----
__device__ __forceinline__ unsigned f2ord(float f) {
    unsigned u = __float_as_uint(f);
    return (u & 0x80000000u) ? ~u : (u | 0x80000000u);
}
__device__ __forceinline__ float ord2f(unsigned e) {
    unsigned b = (e & 0x80000000u) ? (e ^ 0x80000000u) : ~e;
    return __uint_as_float(b);
}

// ---------------------------------------------------------------------------
// Kernel 1: normalize rows of ex/ey -> f16, zero the max accumulators.
// ---------------------------------------------------------------------------
__global__ __launch_bounds__(256) void cossim_norm_kernel(
    const float* __restrict__ ex, const float* __restrict__ ey,
    _Float16* __restrict__ xn, _Float16* __restrict__ yn,
    unsigned* __restrict__ maxenc /* 16384 entries */) {
    int tid = blockIdx.x * 256 + threadIdx.x;
    if (tid < 2 * NROWS) maxenc[tid] = 0u;   // minimum of sortable encoding

    int wave = threadIdx.x >> 5;
    int lane = threadIdx.x & 31;
    int row  = blockIdx.x * 8 + wave;        // 0 .. 16383

    const float* src;
    _Float16*    dst;
    if (row < NROWS) {
        src = ex + (size_t)row * KDIM;
        dst = xn + (size_t)row * KDIM;
    } else {
        src = ey + (size_t)(row - NROWS) * KDIM;
        dst = yn + (size_t)(row - NROWS) * KDIM;
    }

    const float4* s4 = (const float4*)src;
    float4 a = s4[lane * 2 + 0];
    float4 b = s4[lane * 2 + 1];

    float ss = a.x * a.x + a.y * a.y + a.z * a.z + a.w * a.w +
               b.x * b.x + b.y * b.y + b.z * b.z + b.w * b.w;
    ss += __shfl_xor(ss, 1, 32);
    ss += __shfl_xor(ss, 2, 32);
    ss += __shfl_xor(ss, 4, 32);
    ss += __shfl_xor(ss, 8, 32);
    ss += __shfl_xor(ss, 16, 32);

    float inv = 1.0f / fmaxf(sqrtf(ss), 1e-8f);

    h8 o;
    o[0] = (_Float16)(a.x * inv); o[1] = (_Float16)(a.y * inv);
    o[2] = (_Float16)(a.z * inv); o[3] = (_Float16)(a.w * inv);
    o[4] = (_Float16)(b.x * inv); o[5] = (_Float16)(b.y * inv);
    o[6] = (_Float16)(b.z * inv); o[7] = (_Float16)(b.w * inv);
    *(h8*)(dst + lane * 8) = o;
}

// ---------------------------------------------------------------------------
// Kernel 2: LDS-staged WMMA GEMM with fused row/col max.
// Block = 256 threads = 8 waves (4M x 2N), wave tile 32x64 (2x4 WMMA tiles).
// K staged in 4 chunks of 64 via async global->LDS copies, double buffered.
// ---------------------------------------------------------------------------
__device__ __forceinline__ v16h lds_frag(const _Float16* p) {
    h8 lo = *(const h8*)(p);
    h8 hi = *(const h8*)(p + 16);
    return __builtin_shufflevector(lo, hi, 0, 1, 2, 3, 4, 5, 6, 7,
                                           8, 9, 10, 11, 12, 13, 14, 15);
}

__device__ __forceinline__ void async_fill_chunk(
    const _Float16* __restrict__ xn, const _Float16* __restrict__ yn,
    int rowM0, int rowN0, int kc, int wave, int lane) {
    // 256 combined rows (128 A + 128 B), 128 B payload per row-chunk.
    // One async b128 instr moves 512 B = 4 rows; 8 instrs per wave.
    unsigned dbase = (unsigned)((kc & 1) * BUFB);
    int sub  = lane >> 3;          // 0..3 : row within instr
    int col  = (lane & 7) * 8;     // halves within row chunk
#pragma unroll
    for (int i = 0; i < 8; ++i) {
        int idx = wave * 8 + i;          // 0..63
        int row = idx * 4 + sub;         // 0..255 combined
        const _Float16* g = (row < BLK)
            ? (xn + (size_t)(rowM0 + row) * KDIM)
            : (yn + (size_t)(rowN0 + row - BLK) * KDIM);
        g += kc * KCH + col;
        unsigned lds = dbase + (unsigned)row * (ROWH * 2) + (unsigned)col * 2;
        asm volatile("global_load_async_to_lds_b128 %0, %1, off"
                     :: "v"(lds), "v"((const void*)g) : "memory");
    }
}

__global__ __launch_bounds__(256) void cossim_gemm_max_kernel(
    const _Float16* __restrict__ xn, const _Float16* __restrict__ yn,
    unsigned* __restrict__ rmax, unsigned* __restrict__ cmax) {
    extern __shared__ char smem[];
    _Float16* smh = (_Float16*)smem;

    int lane  = threadIdx.x & 31;
    int wave  = threadIdx.x >> 5;
    int waveM = wave & 3;          // 0..3
    int waveN = wave >> 2;         // 0..1
    int rowM0 = blockIdx.y * BLK;
    int rowN0 = blockIdx.x * BLK;
    int baseM = rowM0 + waveM * 32;
    int baseN = rowN0 + waveN * 64;

    int l15  = lane & 15;
    int ksel = (lane >> 4) << 3;   // 0 / 8 per ISA 16-bit A/B lane layout

    v8f acc[2][4];
#pragma unroll
    for (int mi = 0; mi < 2; ++mi)
#pragma unroll
        for (int ni = 0; ni < 4; ++ni) acc[mi][ni] = (v8f){};

    async_fill_chunk(xn, yn, rowM0, rowN0, 0, wave, lane);

    for (int kc = 0; kc < NCHUNK; ++kc) {
        asm volatile("s_wait_asynccnt 0x0" ::: "memory");  // my chunk-kc copies done
        __syncthreads();                                    // everyone's copies done
        if (kc + 1 < NCHUNK)
            async_fill_chunk(xn, yn, rowM0, rowN0, kc + 1, wave, lane);

        const _Float16* smA = smh + (kc & 1) * (BUFB / 2);
        const _Float16* smB = smA + BLK * ROWH;

#pragma unroll
        for (int ks = 0; ks < KCH; ks += 32) {
            v16h fa[2], fb[4];
#pragma unroll
            for (int mi = 0; mi < 2; ++mi) {
                int r = waveM * 32 + mi * 16 + l15;
                fa[mi] = lds_frag(smA + r * ROWH + ks + ksel);
            }
#pragma unroll
            for (int ni = 0; ni < 4; ++ni) {
                int r = waveN * 64 + ni * 16 + l15;
                fb[ni] = lds_frag(smB + r * ROWH + ks + ksel);
            }
#pragma unroll
            for (int mi = 0; mi < 2; ++mi)
#pragma unroll
                for (int ni = 0; ni < 4; ++ni)
                    acc[mi][ni] = __builtin_amdgcn_wmma_f32_16x16x32_f16(
                        false, fa[mi], false, fb[ni], (short)0, acc[mi][ni],
                        false, false);
        }
    }

    // C/D layout: VGPR r, lane l -> element (M = r + 8*(l>=16), N = l & 15).

    // ---- row maxima: reduce over 4 N-tiles then across 16 lanes ----
#pragma unroll
    for (int mi = 0; mi < 2; ++mi) {
        float r[8];
#pragma unroll
        for (int i = 0; i < 8; ++i) {
            r[i] = fmaxf(fmaxf(acc[mi][0][i], acc[mi][1][i]),
                         fmaxf(acc[mi][2][i], acc[mi][3][i]));
        }
#pragma unroll
        for (int i = 0; i < 8; ++i) {
            r[i] = fmaxf(r[i], __shfl_xor(r[i], 1, 32));
            r[i] = fmaxf(r[i], __shfl_xor(r[i], 2, 32));
            r[i] = fmaxf(r[i], __shfl_xor(r[i], 4, 32));
            r[i] = fmaxf(r[i], __shfl_xor(r[i], 8, 32));
        }
        if (l15 == 0) {
            int rb = baseM + mi * 16 + ((lane >> 4) << 3);
#pragma unroll
            for (int i = 0; i < 8; ++i) atomicMax(&rmax[rb + i], f2ord(r[i]));
        }
    }

    // ---- column maxima: reduce 8 regs x 2 M-tiles, then partner lane ----
#pragma unroll
    for (int ni = 0; ni < 4; ++ni) {
        float m = fmaxf(acc[0][ni][0], acc[1][ni][0]);
#pragma unroll
        for (int i = 1; i < 8; ++i)
            m = fmaxf(m, fmaxf(acc[0][ni][i], acc[1][ni][i]));
        m = fmaxf(m, __shfl_xor(m, 16, 32));
        if (lane < 16) atomicMax(&cmax[baseN + ni * 16 + lane], f2ord(m));
    }
}

// ---------------------------------------------------------------------------
// Kernel 3: out = -sum(exp(lp)*lp), lp = -0.5*((-max)/sigma)^2 - ln(s) - 0.5ln(2pi)
// ---------------------------------------------------------------------------
__global__ __launch_bounds__(256) void cossim_finalize_kernel(
    const unsigned* __restrict__ rmax, const unsigned* __restrict__ cmax,
    float* __restrict__ out) {
    __shared__ float sdata[256];
    const float INV_SIGMA = 1.0f / 0.3f;
    const float LPC = 0.28503427f;  // -ln(0.3) - 0.5*ln(2*pi)
    int tid = threadIdx.x;

    float s0 = 0.0f, s1 = 0.0f;
    for (int i = tid; i < NROWS; i += 256) {
        float f  = ord2f(rmax[i]);
        float z  = -f * INV_SIGMA;
        float lp = -0.5f * z * z + LPC;
        s0 += -__expf(lp) * lp;

        float g   = ord2f(cmax[i]);
        float z2  = -g * INV_SIGMA;
        float lp2 = -0.5f * z2 * z2 + LPC;
        s1 += -__expf(lp2) * lp2;
    }

    sdata[tid] = s0;
    __syncthreads();
    for (int s = 128; s > 0; s >>= 1) {
        if (tid < s) sdata[tid] += sdata[tid + s];
        __syncthreads();
    }
    if (tid == 0) out[0] = sdata[0];
    __syncthreads();

    sdata[tid] = s1;
    __syncthreads();
    for (int s = 128; s > 0; s >>= 1) {
        if (tid < s) sdata[tid] += sdata[tid + s];
        __syncthreads();
    }
    if (tid == 0) out[1] = sdata[0];
}

// ---------------------------------------------------------------------------
extern "C" void kernel_launch(void* const* d_in, const int* in_sizes, int n_in,
                              void* d_out, int out_size, void* d_ws, size_t ws_size,
                              hipStream_t stream) {
    const float* ex = (const float*)d_in[0];
    const float* ey = (const float*)d_in[1];
    float* out = (float*)d_out;

    // Workspace: xn f16 (4MB) | yn f16 (4MB) | rmax u32 (32KB) | cmax u32 (32KB)
    char* ws = (char*)d_ws;
    _Float16* xn = (_Float16*)ws;
    _Float16* yn = xn + (size_t)NROWS * KDIM;
    unsigned* maxenc = (unsigned*)(yn + (size_t)NROWS * KDIM);
    unsigned* rmax = maxenc;
    unsigned* cmax = maxenc + NROWS;

    cossim_norm_kernel<<<dim3(2 * NROWS / 8), dim3(256), 0, stream>>>(
        ex, ey, xn, yn, maxenc);

    cossim_gemm_max_kernel<<<dim3(NROWS / BLK, NROWS / BLK), dim3(256),
                             LDSB, stream>>>(xn, yn, rmax, cmax);

    cossim_finalize_kernel<<<dim3(1), dim3(256), 0, stream>>>(rmax, cmax, out);
}
